// emotion_bank_20787641712807
// MI455X (gfx1250) — matrix operand order repaired and checked by reading, not compiled, not run on hardware
//
#include <hip/hip_runtime.h>
#include <hip/hip_bf16.h>

#define IN_DIM   256
#define SRC_DIM  1024
#define CB_K     512
#define N_ROWS   65536   // 16*64*8*8

typedef __attribute__((ext_vector_type(16))) __bf16 v16bf;
typedef __attribute__((ext_vector_type(8)))  __bf16 v8bf;
typedef __attribute__((ext_vector_type(4)))  __bf16 v4bf;
typedef __attribute__((ext_vector_type(2)))  __bf16 v2bf;
typedef __attribute__((ext_vector_type(8)))  float  v8f;

// ---------------------------------------------------------------------------
// float -> bf16 conversion helpers (native packed cvt when clang declares it)
// ---------------------------------------------------------------------------
__device__ __forceinline__ __bf16 f2bf_sw(float f) {
    unsigned u = __builtin_bit_cast(unsigned, f);
    u += 0x7FFFu + ((u >> 16) & 1u);                 // round-to-nearest-even
    unsigned short h = (unsigned short)(u >> 16);
    return __builtin_bit_cast(__bf16, h);
}

#if __has_builtin(__builtin_amdgcn_cvt_pk_bf16_f32)
__device__ __forceinline__ v2bf pk2(float a, float b) {
    return __builtin_amdgcn_cvt_pk_bf16_f32(a, b);
}
__device__ __forceinline__ __bf16 f2bf1(float f) { return pk2(f, f)[0]; }
__device__ __forceinline__ v8bf cvt8(float4 a, float4 b) {
    v2bf p0 = pk2(a.x, a.y), p1 = pk2(a.z, a.w);
    v2bf p2 = pk2(b.x, b.y), p3 = pk2(b.z, b.w);
    v4bf q0 = __builtin_shufflevector(p0, p1, 0, 1, 2, 3);
    v4bf q1 = __builtin_shufflevector(p2, p3, 0, 1, 2, 3);
    return __builtin_shufflevector(q0, q1, 0, 1, 2, 3, 4, 5, 6, 7);
}
#else
__device__ __forceinline__ __bf16 f2bf1(float f) { return f2bf_sw(f); }
__device__ __forceinline__ v8bf cvt8(float4 a, float4 b) {
    v8bf o = { f2bf_sw(a.x), f2bf_sw(a.y), f2bf_sw(a.z), f2bf_sw(a.w),
               f2bf_sw(b.x), f2bf_sw(b.y), f2bf_sw(b.z), f2bf_sw(b.w) };
    return o;
}
#endif

// ---------------------------------------------------------------------------
// P0: elementwise f32 -> bf16 (8 elements / thread), done ONCE per tensor
// ---------------------------------------------------------------------------
__global__ __launch_bounds__(256) void cvt_f32_bf16(const float* __restrict__ src,
                                                    __bf16* __restrict__ dst,
                                                    int n8) {
    int i = blockIdx.x * 256 + threadIdx.x;
    if (i >= n8) return;
    const float4* p = (const float4*)src + (size_t)i * 2;
    float4 a = p[0], b = p[1];
    *((v8bf*)dst + i) = cvt8(a, b);
}

// ---------------------------------------------------------------------------
// P1: codebook squared row norms
// ---------------------------------------------------------------------------
__global__ __launch_bounds__(256) void cbnorm_kernel(const float* __restrict__ cb,
                                                     float* __restrict__ cbnorm) {
    __shared__ float red[8];
    const int k = blockIdx.x;
    const float4* p = (const float4*)(cb + (size_t)k * SRC_DIM);
    float4 v = p[threadIdx.x];                   // 256 float4 per row
    float s = v.x * v.x + v.y * v.y + v.z * v.z + v.w * v.w;
    #pragma unroll
    for (int off = 16; off >= 1; off >>= 1) s += __shfl_down(s, off, 32);
    if ((threadIdx.x & 31) == 0) red[threadIdx.x >> 5] = s;
    __syncthreads();
    if (threadIdx.x == 0) {
        float t = 0.f;
        #pragma unroll
        for (int i = 0; i < 8; ++i) t += red[i];
        cbnorm[k] = t;
    }
}

// concat two 16B bf16 loads into one WMMA operand
__device__ __forceinline__ v16bf cat16(v8bf lo, v8bf hi) {
    return __builtin_shufflevector(lo, hi, 0, 1, 2, 3, 4, 5, 6, 7,
                                           8, 9, 10, 11, 12, 13, 14, 15);
}

__device__ __forceinline__ v16bf wmma_bf16(v16bf A, v16bf B, v8f& C) {
    C = __builtin_amdgcn_wmma_f32_16x16x32_bf16(false, A, false, B,
                                                (short)0, C, false, false);
    return A;
}

// ---------------------------------------------------------------------------
// K1: zp = z @ W_fc^T + b  via v_wmma_f32_16x16x32_bf16, software-pipelined
// block = 128 threads (4 waves); wave -> 16 rows x 64 cols
// grid = (N_ROWS/16, SRC_DIM/256)
// ---------------------------------------------------------------------------
__global__ __launch_bounds__(128) void fc_gemm(const __bf16* __restrict__ zb,
                                               const __bf16* __restrict__ Wb,
                                               const float* __restrict__ bias,
                                               float* __restrict__ zp,
                                               __bf16* __restrict__ zpb) {
    const int lane = threadIdx.x & 31;
    const int wave = threadIdx.x >> 5;
    const int h  = lane >> 4;        // half-wave select
    const int ln = lane & 15;
    const int row0 = blockIdx.x * 16;
    const int col0 = blockIdx.y * 256 + wave * 64;

    v8f acc[4] = {};

    // A lane layout: elems 0..15 -> K = h*16 + i
    const __bf16* aptr = zb + (size_t)(row0 + ln) * IN_DIM + h * 16;
    const __bf16* wptr[4];
    #pragma unroll
    for (int t = 0; t < 4; ++t)
        wptr[t] = Wb + (size_t)(col0 + t * 16 + ln) * IN_DIM + h * 16;

    // prologue: stage kc = 0
    v16bf A = cat16(*(const v8bf*)(aptr), *(const v8bf*)(aptr + 8));
    v16bf B[4];
    #pragma unroll
    for (int t = 0; t < 4; ++t) B[t] = *(const v16bf*)(wptr[t]);

    #pragma unroll
    for (int kc = 0; kc < IN_DIM - 32; kc += 32) {
        // issue next-iteration loads before consuming current tiles
        v16bf An = cat16(*(const v8bf*)(aptr + kc + 32),
                         *(const v8bf*)(aptr + kc + 40));
        v16bf Bn[4];
        #pragma unroll
        for (int t = 0; t < 4; ++t) Bn[t] = *(const v16bf*)(wptr[t] + kc + 32);

        #pragma unroll
        for (int t = 0; t < 4; ++t) wmma_bf16(A, B[t], acc[t]);

        A = An;
        #pragma unroll
        for (int t = 0; t < 4; ++t) B[t] = Bn[t];
    }
    #pragma unroll
    for (int t = 0; t < 4; ++t) wmma_bf16(A, B[t], acc[t]);

    #pragma unroll
    for (int t = 0; t < 4; ++t) {
        const int col = col0 + t * 16 + ln;
        const float bv = bias[col];
        #pragma unroll
        for (int r = 0; r < 8; ++r) {
            const int row = row0 + h * 8 + r;   // D vgpr r -> M = h*8 + r
            const float v = acc[t][r] + bv;
            zp [(size_t)row * SRC_DIM + col] = v;
            zpb[(size_t)row * SRC_DIM + col] = f2bf1(v);
        }
    }
}

// ---------------------------------------------------------------------------
// K2: scores = ||e||^2 - 2 * zp @ cb^T ; argmin over 512 codes
// block = 128 threads (4 waves); wave -> 16 rows x 128 codes/pass, 4 passes
// grid = N_ROWS/64. Pipelined at half-granularity: while WMMA'ing code-group
// g, the loads for group g+1 (and the next A tile) are already in flight.
// ---------------------------------------------------------------------------
__global__ __launch_bounds__(128) void vq_argmin(const __bf16* __restrict__ zpb,
                                                 const __bf16* __restrict__ cbb,
                                                 const float* __restrict__ cbnorm,
                                                 int* __restrict__ idx_out,
                                                 float* __restrict__ idx_f_out) {
    const int lane = threadIdx.x & 31;
    const int wave = threadIdx.x >> 5;
    const int h  = lane >> 4;
    const int ln = lane & 15;
    const int row0 = blockIdx.x * 64 + wave * 16;

    float best[8];
    int   bidx[8];
    #pragma unroll
    for (int r = 0; r < 8; ++r) { best[r] = 3.4e38f; bidx[r] = 0; }

    const __bf16* aptr = zpb + (size_t)(row0 + ln) * SRC_DIM + h * 16;

    for (int pass = 0; pass < 4; ++pass) {
        const int code0 = pass * 128;
        v8f acc[8] = {};

        // prefetch next pass's codebook lines (global_prefetch_b8)
        if (pass < 3) {
            #pragma unroll
            for (int t = 0; t < 8; ++t)
                __builtin_prefetch(cbb + (size_t)(code0 + 128 + t * 16 + ln) * SRC_DIM,
                                   0, 1);
        }

        const __bf16* cptr[8];
        #pragma unroll
        for (int t = 0; t < 8; ++t)
            cptr[t] = cbb + (size_t)(code0 + t * 16 + ln) * SRC_DIM + h * 16;

        // prologue: A(0) and group0 B tiles at kc = 0
        v16bf A = cat16(*(const v8bf*)(aptr), *(const v8bf*)(aptr + 8));
        v16bf B0[4], B1[4];
        #pragma unroll
        for (int u = 0; u < 4; ++u) B0[u] = *(const v16bf*)(cptr[u]);

        for (int kc = 0; kc < SRC_DIM - 32; kc += 32) {
            // in-flight: group1(kc) + A(kc+32)
            #pragma unroll
            for (int u = 0; u < 4; ++u) B1[u] = *(const v16bf*)(cptr[4 + u] + kc);
            v16bf An = cat16(*(const v8bf*)(aptr + kc + 32),
                             *(const v8bf*)(aptr + kc + 40));

            #pragma unroll
            for (int u = 0; u < 4; ++u) wmma_bf16(A, B0[u], acc[u]);

            // in-flight: group0(kc+32)
            #pragma unroll
            for (int u = 0; u < 4; ++u) B0[u] = *(const v16bf*)(cptr[u] + kc + 32);

            #pragma unroll
            for (int u = 0; u < 4; ++u) wmma_bf16(A, B1[u], acc[4 + u]);

            A = An;
        }
        // epilogue kc = SRC_DIM - 32
        #pragma unroll
        for (int u = 0; u < 4; ++u) B1[u] = *(const v16bf*)(cptr[4 + u] + SRC_DIM - 32);
        #pragma unroll
        for (int u = 0; u < 4; ++u) wmma_bf16(A, B0[u], acc[u]);
        #pragma unroll
        for (int u = 0; u < 4; ++u) wmma_bf16(A, B1[u], acc[4 + u]);

        #pragma unroll
        for (int t = 0; t < 8; ++t) {
            const int code = code0 + t * 16 + ln;
            const float cn = cbnorm[code];
            #pragma unroll
            for (int r = 0; r < 8; ++r) {
                const float s = cn - 2.0f * acc[t][r];
                // codes visited ascending per lane -> strict < keeps first min
                if (s < best[r]) { best[r] = s; bidx[r] = code; }
            }
        }
    }

    // reduce across the 16 lanes of each half-wave (xor 1,2,4,8 stays in-half)
    #pragma unroll
    for (int off = 1; off <= 8; off <<= 1) {
        #pragma unroll
        for (int r = 0; r < 8; ++r) {
            const float os = __shfl_xor(best[r], off, 32);
            const int   oi = __shfl_xor(bidx[r], off, 32);
            if (os < best[r] || (os == best[r] && oi < bidx[r])) {
                best[r] = os; bidx[r] = oi;
            }
        }
    }

    if (ln == 0) {
        #pragma unroll
        for (int r = 0; r < 8; ++r) {
            const int row = row0 + h * 8 + r;
            idx_out[row]   = bidx[r];
            idx_f_out[row] = (float)bidx[r];
        }
    }
}

// ---------------------------------------------------------------------------
// K3: quantized[row] = cb[idx[row]] ; partial[row] = sum((q - zp)^2)
// grid = N_ROWS, block = 256 (one float4 per thread)
// ---------------------------------------------------------------------------
__global__ __launch_bounds__(256) void gather_loss(const float* __restrict__ zp,
                                                   const float* __restrict__ cb,
                                                   const int* __restrict__ idx,
                                                   float* __restrict__ qout,
                                                   float* __restrict__ partial) {
    __shared__ float red[8];
    const int row = blockIdx.x;
    const int k = idx[row];
    const int t = threadIdx.x;
    const float4* cp  = (const float4*)(cb + (size_t)k   * SRC_DIM);
    const float4* zpp = (const float4*)(zp + (size_t)row * SRC_DIM);
    float4*       qp  = (float4*)(qout + (size_t)row * SRC_DIM);
    float4 q = cp[t];
    float4 z = zpp[t];
    qp[t] = q;
    const float dx = q.x - z.x, dy = q.y - z.y, dz = q.z - z.z, dw = q.w - z.w;
    float s = dx * dx + dy * dy + dz * dz + dw * dw;
    #pragma unroll
    for (int off = 16; off >= 1; off >>= 1) s += __shfl_down(s, off, 32);
    if ((threadIdx.x & 31) == 0) red[threadIdx.x >> 5] = s;
    __syncthreads();
    if (threadIdx.x == 0) {
        float tt = 0.f;
        #pragma unroll
        for (int i = 0; i < 8; ++i) tt += red[i];
        partial[row] = tt;
    }
}

// ---------------------------------------------------------------------------
// K4: deterministic final reduction; loss = (1 + 0.25) * mean((q - zp)^2)
// ---------------------------------------------------------------------------
__global__ __launch_bounds__(1024) void finalize_loss(const float* __restrict__ partial,
                                                      float* __restrict__ loss_out) {
    __shared__ float red[32];
    float s = 0.f;
    for (int i = threadIdx.x; i < N_ROWS; i += 1024) s += partial[i];
    #pragma unroll
    for (int off = 16; off >= 1; off >>= 1) s += __shfl_down(s, off, 32);
    if ((threadIdx.x & 31) == 0) red[threadIdx.x >> 5] = s;
    __syncthreads();
    if (threadIdx.x == 0) {
        float t = 0.f;
        #pragma unroll
        for (int i = 0; i < 32; ++i) t += red[i];
        loss_out[0] = 1.25f * t / (float)((size_t)N_ROWS * SRC_DIM);
    }
}

// ---------------------------------------------------------------------------
extern "C" void kernel_launch(void* const* d_in, const int* in_sizes, int n_in,
                              void* d_out, int out_size, void* d_ws, size_t ws_size,
                              hipStream_t stream) {
    (void)in_sizes; (void)n_in; (void)out_size; (void)ws_size;

    const float* z   = (const float*)d_in[0];   // [16,64,8,8,256]
    const float* Wfc = (const float*)d_in[1];   // [1024,256]
    const float* bfc = (const float*)d_in[2];   // [1024]
    const float* cb  = (const float*)d_in[3];   // [512,1024]

    char* ws = (char*)d_ws;
    size_t off = 0;
    float* zp = (float*)(ws + off);  off += (size_t)N_ROWS * SRC_DIM * sizeof(float);
    __bf16* zpb = (__bf16*)(ws + off); off += (size_t)N_ROWS * SRC_DIM * sizeof(__bf16);
    __bf16* zb  = (__bf16*)(ws + off); off += (size_t)N_ROWS * IN_DIM * sizeof(__bf16);
    __bf16* Wb  = (__bf16*)(ws + off); off += (size_t)SRC_DIM * IN_DIM * sizeof(__bf16);
    __bf16* cbb = (__bf16*)(ws + off); off += (size_t)CB_K * SRC_DIM * sizeof(__bf16);
    float* cbnorm  = (float*)(ws + off); off += (size_t)CB_K * sizeof(float);
    int*   idx     = (int*)  (ws + off); off += (size_t)N_ROWS * sizeof(int);
    float* partial = (float*)(ws + off);

    float* qout  = (float*)d_out;                         // [N,1024]
    float* lossp = qout + (size_t)N_ROWS * SRC_DIM;       // scalar
    float* idxf  = lossp + 1;                             // [N,1] as float

    // one-time bf16 conversions
    {
        int n8 = (N_ROWS * IN_DIM) / 8;                   // 2,097,152
        cvt_f32_bf16<<<n8 / 256, 256, 0, stream>>>(z, zb, n8);
    }
    {
        int n8 = (SRC_DIM * IN_DIM) / 8;                  // 32,768
        cvt_f32_bf16<<<n8 / 256, 256, 0, stream>>>(Wfc, Wb, n8);
    }
    {
        int n8 = (CB_K * SRC_DIM) / 8;                    // 65,536
        cvt_f32_bf16<<<n8 / 256, 256, 0, stream>>>(cb, cbb, n8);
    }

    cbnorm_kernel<<<CB_K, 256, 0, stream>>>(cb, cbnorm);

    dim3 g1(N_ROWS / 16, SRC_DIM / 256);
    fc_gemm<<<g1, 128, 0, stream>>>(zb, Wb, bfc, zp, zpb);

    vq_argmin<<<N_ROWS / 64, 128, 0, stream>>>(zpb, cbb, cbnorm, idx, idxf);

    gather_loss<<<N_ROWS, 256, 0, stream>>>(zp, cb, idx, qout, partial);

    finalize_loss<<<1, 1024, 0, stream>>>(partial, lossp);
}
